// GRADE_LP_22995254903269
// MI455X (gfx1250) — compile-verified, matrix operand order checked by reading, not run on hardware
//
#include <hip/hip_runtime.h>

#define NN     110000
#define FDIM   128
#define NUSER  60000

typedef float  v2f   __attribute__((ext_vector_type(2)));
typedef float  v8f   __attribute__((ext_vector_type(8)));
typedef __bf16 v16bf __attribute__((ext_vector_type(16)));

__device__ __forceinline__ float atomAddF(float* p, float v) {
    return unsafeAtomicAdd(p, v);
}

// BCE term, torch BCELoss semantics (log clamped at -100)
__device__ __forceinline__ float bce_term(float z, float y) {
    float p  = 1.f / (1.f + __expf(-z));
    float lp = fmaxf(__logf(p), -100.f);
    float lq = fmaxf(__logf(1.f - p), -100.f);
    return -(y * lp + (1.f - y) * lq);
}

// ---------------------------------------------------------------- utilities
__global__ void k_zero1(float* p) { if (threadIdx.x == 0 && blockIdx.x == 0) p[0] = 0.f; }

__global__ void k_fill1(float* deg, int n) {
    int i = blockIdx.x * blockDim.x + threadIdx.x;
    if (i < n) deg[i] = 1.0f;          // self-loop contributes 1 to every degree
}

__global__ void k_deg_edges(const int* __restrict__ dst, float* __restrict__ deg, int E) {
    int e = blockIdx.x * blockDim.x + threadIdx.x;
    if (e < E) atomAddF(&deg[dst[e]], 1.0f);
}

__global__ void k_rsqrt(float* d, int n) {
    int i = blockIdx.x * blockDim.x + threadIdx.x;
    if (i < n) d[i] = __frsqrt_rn(d[i]);   // deg >= 1 always
}

// ---------------------------------------------------------------- WMMA GEMM
// C[110000x128] = X[110000x128] @ W[128x128], row-major fp32.
// One wave per 16x16 output tile; 8 waves/block cover all 8 N-tiles of one M-tile.
__global__ void __launch_bounds__(256)
k_gemm(const float* __restrict__ X, const float* __restrict__ W, float* __restrict__ C) {
    const int lane = threadIdx.x & 31;
    const int wave = threadIdx.x >> 5;
    const int m0   = blockIdx.x * 16;       // 110000/16 = 6875 exact
    const int n0   = wave * 16;
    const int row  = lane & 15;             // M index (A) / N index (B,C) in tile
    const int half = lane >> 4;
    const float* Xrow = X + (size_t)(m0 + row) * FDIM;
    v8f acc = {};
#if __has_builtin(__builtin_amdgcn_wmma_f32_16x16x4_f32)
    // fp32 WMMA: A 16x4 (v2f: lanes0-15 K=k,k+1; lanes16-31 K=k+2,k+3), B 4x16 analogous.
    const v2f* X2 = (const v2f*)Xrow;
    #pragma unroll
    for (int k = 0; k < FDIM; k += 4) {
        const int kk = k + 2 * half;
        v2f a = X2[kk >> 1];
        v2f b;
        b.x = W[(size_t)kk * FDIM + n0 + row];
        b.y = W[(size_t)(kk + 1) * FDIM + n0 + row];
        acc = __builtin_amdgcn_wmma_f32_16x16x4_f32(false, a, false, b,
                                                    (short)0, acc, false, false);
    }
#else
    // Fallback: bf16 WMMA 16x16x32 (codegen-confirmed builtin), ISA 16-bit A layout.
    #pragma unroll
    for (int k = 0; k < FDIM; k += 32) {
        v16bf a, b;
        #pragma unroll
        for (int j = 0; j < 8; ++j) {
            int kk = k + ((j < 4) ? (8 * half + 2 * j) : (16 + 8 * half + 2 * (j - 4)));
            a[2 * j]     = (__bf16)Xrow[kk];
            a[2 * j + 1] = (__bf16)Xrow[kk + 1];
            b[2 * j]     = (__bf16)W[(size_t)kk * FDIM + n0 + row];
            b[2 * j + 1] = (__bf16)W[(size_t)(kk + 1) * FDIM + n0 + row];
        }
        acc = __builtin_amdgcn_wmma_f32_16x16x32_bf16(false, a, false, b,
                                                      (short)0, acc, false, false);
    }
#endif
    const int mb = m0 + half * 8;           // C layout: VGPR r -> M = r (+8 for hi half)
    #pragma unroll
    for (int r = 0; r < 8; ++r)
        C[(size_t)(mb + r) * FDIM + n0 + row] = acc[r];
}

// ------------------------------------------------- aggregation (A+I normalized)
// Init agg with self-loop term: agg[i] = xw[i] * dinv[i]^2   (one float4 / thread)
__global__ void k_selfloop(const float* __restrict__ xw, const float* __restrict__ dinv,
                           float* __restrict__ agg) {
    int i = blockIdx.x * blockDim.x + threadIdx.x;   // over NN*32 float4 units
    if (i >= NN * (FDIM / 4)) return;
    int   node = i >> 5;
    float di   = dinv[node];
    float w    = di * di;
    float4 v = ((const float4*)xw)[i];
    float4 o; o.x = v.x * w; o.y = v.y * w; o.z = v.z * w; o.w = v.w * w;
    ((float4*)agg)[i] = o;
}

// One wave per edge: gather xw[src] row (float4/lane), scale, atomic-scatter to agg[dst].
__global__ void __launch_bounds__(256)
k_edge_agg(const float* __restrict__ xw, const int* __restrict__ src,
           const int* __restrict__ dst, const float* __restrict__ dinv,
           float* __restrict__ agg, int E) {
    int e = blockIdx.x * 8 + (threadIdx.x >> 5);
    if (e >= E) return;
    int lane = threadIdx.x & 31;
    if (lane == 0 && e + 2048 < E) {
        __builtin_prefetch(src + e + 2048, 0, 1);    // global_prefetch_b8
        __builtin_prefetch(dst + e + 2048, 0, 1);
    }
    int   s = src[e], d = dst[e];
    float w = dinv[s] * dinv[d];
    float4 v = ((const float4*)(xw + (size_t)s * FDIM))[lane];
    float* o = agg + (size_t)d * FDIM + lane * 4;
    atomAddF(o + 0, v.x * w);
    atomAddF(o + 1, v.y * w);
    atomAddF(o + 2, v.z * w);
    atomAddF(o + 3, v.w * w);
}

// h = relu(agg + bias)  (one float4 / thread)
__global__ void k_bias_relu(const float* __restrict__ agg, const float* __restrict__ b,
                            float* __restrict__ h) {
    int i = blockIdx.x * blockDim.x + threadIdx.x;
    if (i >= NN * (FDIM / 4)) return;
    int c = (i & 31) * 4;
    float4 v  = ((const float4*)agg)[i];
    float4 bb = *(const float4*)(b + c);
    float4 o;
    o.x = fmaxf(v.x + bb.x, 0.f); o.y = fmaxf(v.y + bb.y, 0.f);
    o.z = fmaxf(v.z + bb.z, 0.f); o.w = fmaxf(v.w + bb.w, 0.f);
    ((float4*)h)[i] = o;
}

// ------------------------------------------------------------- loss heads
// Prediction head: sigmoid([h1[u],h2[u],h1[i],h2[i]] . Wp + bp), BCE vs labels.
__global__ void __launch_bounds__(256)
k_pred_loss(const float* __restrict__ h1, const float* __restrict__ h2,
            const int* __restrict__ user, const int* __restrict__ item,
            const int* __restrict__ labels, const float* __restrict__ Wp,
            const float* __restrict__ bp, float* acc, int B, float scale) {
    int r = blockIdx.x * 8 + (threadIdx.x >> 5);
    if (r >= B) return;
    int lane = threadIdx.x & 31;
    int u  = user[r];
    int it = item[r] + NUSER;
    const float4* a0 = (const float4*)(h1 + (size_t)u  * FDIM);
    const float4* a1 = (const float4*)(h2 + (size_t)u  * FDIM);
    const float4* a2 = (const float4*)(h1 + (size_t)it * FDIM);
    const float4* a3 = (const float4*)(h2 + (size_t)it * FDIM);
    const float4* w4 = (const float4*)Wp;          // 512 floats = 128 float4
    float4 x, w;
    float s = 0.f;
    x = a0[lane]; w = w4[lane];      s += x.x*w.x + x.y*w.y + x.z*w.z + x.w*w.w;
    x = a1[lane]; w = w4[32 + lane]; s += x.x*w.x + x.y*w.y + x.z*w.z + x.w*w.w;
    x = a2[lane]; w = w4[64 + lane]; s += x.x*w.x + x.y*w.y + x.z*w.z + x.w*w.w;
    x = a3[lane]; w = w4[96 + lane]; s += x.x*w.x + x.y*w.y + x.z*w.z + x.w*w.w;
    #pragma unroll
    for (int m = 16; m; m >>= 1) s += __shfl_xor(s, m, 32);
    if (lane == 0)
        atomAddF(acc, bce_term(s + bp[0], (float)labels[r]) * scale);
}

// Domain head: rows [0,B) from domain A (label 0), [B,2B) from domain B (label 1).
__global__ void __launch_bounds__(256)
k_domain_loss(const float* __restrict__ h1a, const float* __restrict__ h2a,
              const int* __restrict__ idxa,
              const float* __restrict__ h1b, const float* __restrict__ h2b,
              const int* __restrict__ idxb, int offset,
              const float* __restrict__ Wd, const float* __restrict__ bd,
              float* acc, int B, float scale) {
    int r = blockIdx.x * 8 + (threadIdx.x >> 5);
    if (r >= 2 * B) return;
    int lane = threadIdx.x & 31;
    const float *h1, *h2; int idx; float y;
    if (r < B) { h1 = h1a; h2 = h2a; idx = idxa[r];     y = 0.f; }
    else       { h1 = h1b; h2 = h2b; idx = idxb[r - B]; y = 1.f; }
    idx += offset;
    const float4* f1 = (const float4*)(h1 + (size_t)idx * FDIM);
    const float4* f2 = (const float4*)(h2 + (size_t)idx * FDIM);
    const float4* w4 = (const float4*)Wd;          // 256 floats = 64 float4
    float4 a = f1[lane], b = f2[lane], wa = w4[lane], wb = w4[32 + lane];
    float s = a.x*wa.x + a.y*wa.y + a.z*wa.z + a.w*wa.w
            + b.x*wb.x + b.y*wb.y + b.z*wb.z + b.w*wb.w;
    #pragma unroll
    for (int m = 16; m; m >>= 1) s += __shfl_xor(s, m, 32);
    if (lane == 0)
        atomAddF(acc, bce_term(s + bd[0], y) * scale);
}

// ---------------------------------------------------------------- launcher
extern "C" void kernel_launch(void* const* d_in, const int* in_sizes, int n_in,
                              void* d_out, int out_size, void* d_ws, size_t ws_size,
                              hipStream_t stream) {
    const float* feats_s = (const float*)d_in[0];
    const float* feats_t = (const float*)d_in[1];
    const int*   adj_s   = (const int*)d_in[2];
    const int*   adj_t   = (const int*)d_in[3];
    const int*   user_s  = (const int*)d_in[4];
    const int*   item_s  = (const int*)d_in[5];
    const int*   lab_s   = (const int*)d_in[6];
    const int*   user_t  = (const int*)d_in[7];
    const int*   item_t  = (const int*)d_in[8];
    const int*   lab_t   = (const int*)d_in[9];
    const float* W1  = (const float*)d_in[10];
    const float* b1  = (const float*)d_in[11];
    const float* W2  = (const float*)d_in[12];
    const float* b2  = (const float*)d_in[13];
    const float* Wp  = (const float*)d_in[14];
    const float* bp  = (const float*)d_in[15];
    const float* Wdu = (const float*)d_in[16];
    const float* bdu = (const float*)d_in[17];
    const float* Wdi = (const float*)d_in[18];
    const float* bdi = (const float*)d_in[19];
    const int E1 = in_sizes[2] / 2;
    const int E2 = in_sizes[3] / 2;
    const int B  = in_sizes[4];
    float* out = (float*)d_out;

    // workspace layout (floats)
    const size_t NNF = (size_t)NN * FDIM;
    float* ws   = (float*)d_ws;
    float* xw   = ws;
    float* agg  = ws + NNF;
    float* h1s  = ws + 2 * NNF;
    float* h2s  = ws + 3 * NNF;
    float* h1t  = ws + 4 * NNF;
    float* h2t  = ws + 5 * NNF;
    float* dinv = ws + 6 * NNF;          // NN floats, recomputed per domain

    const int gNN  = (NN + 255) / 256;
    const int gNF4 = (NN * (FDIM / 4) + 255) / 256;   // float4 grid over NN*128

    k_zero1<<<1, 1, 0, stream>>>(out);

    auto run_domain = [&](const float* feats, const int* adj, int E,
                          float* h1, float* h2) {
        const int* src = adj;
        const int* dst = adj + E;
        const int gE  = (E + 255) / 256;
        const int gEw = (E + 7) / 8;     // one wave per edge, 8 waves/block
        // degrees -> dinv
        k_fill1    <<<gNN, 256, 0, stream>>>(dinv, NN);
        k_deg_edges<<<gE,  256, 0, stream>>>(dst, dinv, E);
        k_rsqrt    <<<gNN, 256, 0, stream>>>(dinv, NN);
        // layer 1
        k_gemm     <<<NN / 16, 256, 0, stream>>>(feats, W1, xw);
        k_selfloop <<<gNF4, 256, 0, stream>>>(xw, dinv, agg);
        k_edge_agg <<<gEw,  256, 0, stream>>>(xw, src, dst, dinv, agg, E);
        k_bias_relu<<<gNF4, 256, 0, stream>>>(agg, b1, h1);
        // layer 2
        k_gemm     <<<NN / 16, 256, 0, stream>>>(h1, W2, xw);
        k_selfloop <<<gNF4, 256, 0, stream>>>(xw, dinv, agg);
        k_edge_agg <<<gEw,  256, 0, stream>>>(xw, src, dst, dinv, agg, E);
        k_bias_relu<<<gNF4, 256, 0, stream>>>(agg, b2, h2);
    };

    run_domain(feats_s, adj_s, E1, h1s, h2s);
    run_domain(feats_t, adj_t, E2, h1t, h2t);

    const int gB  = (B + 7) / 8;
    const int gB2 = (2 * B + 7) / 8;
    const float invB  = 1.0f / (float)B;
    const float dscal = 0.1f / (2.0f * (float)B);

    k_pred_loss<<<gB, 256, 0, stream>>>(h1s, h2s, user_s, item_s, lab_s, Wp, bp, out, B, invB);
    k_pred_loss<<<gB, 256, 0, stream>>>(h1t, h2t, user_t, item_t, lab_t, Wp, bp, out, B, invB);
    // domain losses (gradient reversal is identity in forward)
    k_domain_loss<<<gB2, 256, 0, stream>>>(h1s, h2s, user_s, h1t, h2t, user_t,
                                           0,     Wdu, bdu, out, B, dscal);
    k_domain_loss<<<gB2, 256, 0, stream>>>(h1s, h2s, item_s, h1t, h2t, item_t,
                                           NUSER, Wdi, bdi, out, B, dscal);
}
